// Model_3805341024655
// MI455X (gfx1250) — matrix operand order repaired and checked by reading, not compiled, not run on hardware
//
#include <hip/hip_runtime.h>
#include <math.h>

typedef _Float16 f16;
typedef __attribute__((ext_vector_type(16))) _Float16 v16h;
typedef __attribute__((ext_vector_type(8)))  _Float16 v8h;
typedef __attribute__((ext_vector_type(8)))  float    v8f;

static __device__ __forceinline__ float leakyf(float x) {
    return x >= 0.f ? x : 0.01f * x;
}

// ---------------- weight packing (f32 -> padded f16) ----------------
// conv weights (O, Ktot) contiguous -> [Op][Kp] zero-padded
__global__ void k_pack_w(const float* __restrict__ s, f16* __restrict__ d,
                         int O, int Kt, int Op, int Kp) {
    int i = blockIdx.x * blockDim.x + threadIdx.x;
    int n = Op * Kp;
    if (i >= n) return;
    int o = i / Kp, k = i % Kp;
    d[i] = (o < O && k < Kt) ? (f16)s[o * Kt + k] : (f16)0.f;
}

// fc weights (K, N) -> transposed padded [Np][Kp]
__global__ void k_pack_wt(const float* __restrict__ s, f16* __restrict__ d,
                          int K, int N, int Np, int Kp) {
    int i = blockIdx.x * blockDim.x + threadIdx.x;
    int tot = Np * Kp;
    if (i >= tot) return;
    int nn = i / Kp, k = i % Kp;
    d[i] = (nn < N && k < K) ? (f16)s[k * N + nn] : (f16)0.f;
}

// caps_w (40, 11, 220) -> [40][224][32]  ( [i][n][d] )
__global__ void k_pack_caps(const float* __restrict__ s, f16* __restrict__ d) {
    int i = blockIdx.x * blockDim.x + threadIdx.x;
    const int tot = 40 * 224 * 32;
    if (i >= tot) return;
    int ci = i / (224 * 32);
    int r  = i % (224 * 32);
    int nn = r / 32, dd = r % 32;
    d[i] = (nn < 220 && dd < 11) ? (f16)s[(ci * 11 + dd) * 220 + nn] : (f16)0.f;
}

__global__ void k_zero16(f16* __restrict__ d, int n) {
    int i = blockIdx.x * blockDim.x + threadIdx.x;
    if (i < n) d[i] = (f16)0.f;
}

// ---------------- im2col into [Lout][B][Kp], K zero-padded ----------------
__global__ void k_im2col_f16(const f16* __restrict__ X, f16* __restrict__ IM,
                             int Bn, int Cin, int Lin, int KW, int Lout,
                             int Kt, int Kp) {
    int i = blockIdx.x * blockDim.x + threadIdx.x;
    int n = Lout * Bn * Kp;
    if (i >= n) return;
    int k = i % Kp;
    int t = i / Kp;
    int b = t % Bn;
    int l = t / Bn;
    f16 v = (f16)0.f;
    if (k < Kt) {
        int ci = k / KW, kw = k - ci * KW;
        v = X[((size_t)b * Cin + ci) * Lin + l + kw];
    }
    IM[i] = v;
}

__global__ void k_im2col_f32(const float* __restrict__ X, f16* __restrict__ IM,
                             int Bn, int Cin, int Lin, int KW, int Lout,
                             int Kt, int Kp) {
    int i = blockIdx.x * blockDim.x + threadIdx.x;
    int n = Lout * Bn * Kp;
    if (i >= n) return;
    int k = i % Kp;
    int t = i / Kp;
    int b = t % Bn;
    int l = t / Bn;
    f16 v = (f16)0.f;
    if (k < Kt) {
        int ci = k / KW, kw = k - ci * KW;
        v = (f16)X[((size_t)b * Cin + ci) * Lin + l + kw];
    }
    IM[i] = v;
}

// ---------------- unified WMMA GEMM, NT output tiles per wave -----------
// A: [z][M][KP] f16 (K-contiguous, KP mult of 32, zero-padded)
// B: [z][Npad][KP] f16 (K-contiguous per output column, padded -> no guards)
// C element address: row*ldcRow + cOff + col*ldcCol + z*cStep
// MODE 0: raw store (skip col>=N)
// MODE 1: bias+leaky (zero-store col>=N to build padded activations)
// MODE 2: conv bias + eval-BN + leaky (skip col>=N)
template <int KP, int NT, int MODE>
__global__ __launch_bounds__(32)
void k_gemm16(const f16* __restrict__ A, long long aStep, int lda,
              const f16* __restrict__ Bw, long long bStep,
              const float* __restrict__ p0, const float* __restrict__ p1,
              const float* __restrict__ p2,
              f16* __restrict__ C, long long cStep, int ldcRow, int ldcCol,
              int cOff, int N)
{
    const int lane = threadIdx.x;
    const int half = lane >> 4, lr = lane & 15;
    const int row  = blockIdx.x * 16 + lr;
    const int z    = blockIdx.z;
    const int col0 = blockIdx.y * (16 * NT) + lr;

    const f16* Ap = A + (size_t)z * aStep + (size_t)row * lda;
    const f16* Bp = Bw + (size_t)z * bStep + (size_t)col0 * KP;

    const v8f zero = {0.f, 0.f, 0.f, 0.f, 0.f, 0.f, 0.f, 0.f};
    v8f acc[NT];
#pragma unroll
    for (int t = 0; t < NT; ++t) acc[t] = zero;

#pragma unroll
    for (int kt = 0; kt < KP / 32; ++kt) {
        // A 16x32 f16 fragment: two contiguous 16B chunks per lane.
        v8h alo = *(const v8h*)(Ap + kt * 32 + half * 8);
        v8h ahi = *(const v8h*)(Ap + kt * 32 + half * 8 + 16);
        v16h a = __builtin_shufflevector(alo, ahi, 0, 1, 2, 3, 4, 5, 6, 7,
                                         8, 9, 10, 11, 12, 13, 14, 15);
#pragma unroll
        for (int t = 0; t < NT; ++t) {
            // B 32x16 fragment: K contiguous per column -> two 16B chunks, same k order as A.
            v8h blo = *(const v8h*)(Bp + (size_t)t * 16 * KP + kt * 32 + half * 8);
            v8h bhi = *(const v8h*)(Bp + (size_t)t * 16 * KP + kt * 32 + half * 8 + 16);
            v16h b = __builtin_shufflevector(blo, bhi, 0, 1, 2, 3, 4, 5, 6, 7,
                                             8, 9, 10, 11, 12, 13, 14, 15);
            acc[t] = __builtin_amdgcn_wmma_f32_16x16x32_f16(
                         false, a, false, b, (short)0, acc[t], false, false);
        }
    }

#pragma unroll
    for (int t = 0; t < NT; ++t) {
        const int col = col0 + t * 16;
        if (col < N) {
            float cb = 0.f, sc = 1.f, bt = 0.f;
            if (MODE == 2) {
                cb = p0[col];
                sc = p1[col] * rsqrtf(1.f + 1e-5f);
                bt = p2[col];
            } else if (MODE == 1) {
                cb = p0[col];
            }
#pragma unroll
            for (int r = 0; r < 8; ++r) {
                int ri = blockIdx.x * 16 + r + half * 8;
                float v = acc[t][r];
                if (MODE == 2)      v = leakyf((v + cb) * sc + bt);
                else if (MODE == 1) v = leakyf(v + cb);
                C[(size_t)ri * ldcRow + cOff + col * ldcCol + (size_t)z * cStep] = (f16)v;
            }
        } else if (MODE == 1) {
#pragma unroll
            for (int r = 0; r < 8; ++r) {
                int ri = blockIdx.x * 16 + r + half * 8;
                C[(size_t)ri * ldcRow + cOff + col * ldcCol + (size_t)z * cStep] = (f16)0.f;
            }
        }
    }
}

// ---------------- per-position attention (softmax over S) ----------------
__global__ __launch_bounds__(32)
void k_attention(const f16* __restrict__ X, const float* __restrict__ w,
                 const float* __restrict__ bias, f16* __restrict__ Y,
                 int C, int S)
{
    const int b = blockIdx.x;
    const int lane = threadIdx.x;
    const f16* x = X + (size_t)b * C * S;

    float a = -1e30f;
    if (lane < S) {
        float dot = bias[lane];
        for (int c = 0; c < C; ++c)
            dot += (float)x[(size_t)c * S + lane] * w[(size_t)lane * C + c];
        a = dot;
    }
    float m = a;
#pragma unroll
    for (int off = 16; off > 0; off >>= 1)
        m = fmaxf(m, __shfl_xor(m, off, 32));
    float e = (lane < S) ? __expf(a - m) : 0.f;
    float sum = e;
#pragma unroll
    for (int off = 16; off > 0; off >>= 1)
        sum += __shfl_xor(sum, off, 32);

    __shared__ float as_[32];
    as_[lane] = e / sum;
    __syncthreads();

    f16* y = Y + (size_t)b * C * S;
    for (int idx = lane; idx < C * S; idx += 32) {
        int s = idx % S;
        y[idx] = (f16)((float)x[idx] * as_[s]);
    }
}

// ---------------- maxpool k=2 (uses first 26 of 27) ----------------
__global__ void k_maxpool(const f16* __restrict__ X, f16* __restrict__ Y,
                          int n, int Lin, int Lo)
{
    int i = blockIdx.x * blockDim.x + threadIdx.x;
    if (i >= n) return;
    int j = i % Lo;
    int bc = i / Lo;
    float a = (float)X[(size_t)bc * Lin + 2 * j];
    float c = (float)X[(size_t)bc * Lin + 2 * j + 1];
    Y[i] = (f16)fmaxf(a, c);
}

// ---------------- dynamic routing (3 iters), one sample per workgroup ----
__global__ __launch_bounds__(256)
void k_routing(const f16* __restrict__ UP, f16* __restrict__ CTX)
{
    __shared__ float u[40 * 220];
    __shared__ float sv[220], vv[220], coef[20];
    __shared__ float bc[800], cc[800];
    const int b = blockIdx.x, tid = threadIdx.x;
    const f16* up = UP + (size_t)b * 40 * 224;

    for (int t = tid; t < 8800; t += 256) {
        int i = t / 220, n = t % 220;
        u[t] = (float)up[(size_t)i * 224 + n];
    }
    for (int t = tid; t < 800; t += 256) bc[t] = 0.f;
    __syncthreads();

    for (int t = tid; t < 220; t += 256) {
        float s = 0.f;
        for (int i = 0; i < 40; ++i) s += u[i * 220 + t];
        sv[t] = s * (1.f / 20.f);
    }
    __syncthreads();
    if (tid < 20) {
        float l2 = 1e-7f;
        for (int d = 0; d < 11; ++d) { float x = sv[tid * 11 + d]; l2 += x * x; }
        coef[tid] = l2 / ((l2 + 1.f) * sqrtf(l2));
    }
    __syncthreads();
    for (int t = tid; t < 220; t += 256) vv[t] = sv[t] * coef[t / 11];
    __syncthreads();

    for (int it = 0; it < 3; ++it) {
        for (int t = tid; t < 800; t += 256) {
            int i = t / 20, o = t % 20;
            float dot = 0.f;
            for (int d = 0; d < 11; ++d)
                dot += u[i * 220 + o * 11 + d] * vv[o * 11 + d];
            bc[t] += dot;
        }
        __syncthreads();
        if (tid < 40) {
            float m = -1e30f;
            for (int o = 0; o < 20; ++o) m = fmaxf(m, bc[tid * 20 + o]);
            float sum = 0.f;
            for (int o = 0; o < 20; ++o) {
                float e = __expf(bc[tid * 20 + o] - m);
                cc[tid * 20 + o] = e; sum += e;
            }
            float inv = 1.f / sum;
            for (int o = 0; o < 20; ++o) cc[tid * 20 + o] *= inv;
        }
        __syncthreads();
        for (int t = tid; t < 220; t += 256) {
            int o = t / 11;
            float s = 0.f;
            for (int i = 0; i < 40; ++i) s += cc[i * 20 + o] * u[i * 220 + t];
            sv[t] = s;
        }
        __syncthreads();
        if (tid < 20) {
            float l2 = 1e-7f;
            for (int d = 0; d < 11; ++d) { float x = sv[tid * 11 + d]; l2 += x * x; }
            coef[tid] = l2 / ((l2 + 1.f) * sqrtf(l2));
        }
        __syncthreads();
        for (int t = tid; t < 220; t += 256) vv[t] = sv[t] * coef[t / 11];
        __syncthreads();
    }

    for (int t = tid; t < 224; t += 256)
        CTX[(size_t)b * 224 + t] = (t < 220) ? (f16)vv[t] : (f16)0.f;
}

// ---------------- final fc3 (N=1) + sigmoid ----------------
__global__ void k_fc3(const f16* __restrict__ H, const float* __restrict__ w,
                      const float* __restrict__ bias, float* __restrict__ out,
                      int Bn)
{
    int i = blockIdx.x * blockDim.x + threadIdx.x;
    if (i >= Bn) return;
    float z = bias[0];
    const v8h* h = (const v8h*)(H + (size_t)i * 224);
    for (int k8 = 0; k8 < 25; ++k8) {
        v8h hv = h[k8];
#pragma unroll
        for (int j = 0; j < 8; ++j) z += (float)hv[j] * w[k8 * 8 + j];
    }
    out[i] = 1.f / (1.f + __expf(-z));
}

// =====================================================================
extern "C" void kernel_launch(void* const* d_in, const int* in_sizes, int n_in,
                              void* d_out, int out_size, void* d_ws, size_t ws_size,
                              hipStream_t stream)
{
    (void)n_in; (void)out_size; (void)ws_size;
    const float* x2    = (const float*)d_in[2];   // hla_a_seqs2 (B,23,34)
    const float* pep   = (const float*)d_in[4];   // peptides    (B,23,15)
    const float* c1w   = (const float*)d_in[8];
    const float* c1b   = (const float*)d_in[9];
    const float* bn1g  = (const float*)d_in[10];
    const float* bn1b  = (const float*)d_in[11];
    const float* c12w  = (const float*)d_in[12];
    const float* c12b  = (const float*)d_in[13];
    const float* bn12g = (const float*)d_in[14];
    const float* bn12b = (const float*)d_in[15];
    const float* atthw = (const float*)d_in[16];
    const float* atthb = (const float*)d_in[17];
    const float* c2w   = (const float*)d_in[18];
    const float* c2b   = (const float*)d_in[19];
    const float* bn2g  = (const float*)d_in[20];
    const float* bn2b  = (const float*)d_in[21];
    const float* c3w   = (const float*)d_in[22];
    const float* c3b   = (const float*)d_in[23];
    const float* bn3g  = (const float*)d_in[24];
    const float* bn3b  = (const float*)d_in[25];
    const float* p0w   = (const float*)d_in[26];
    const float* attpw = (const float*)d_in[27];
    const float* attpb = (const float*)d_in[28];
    const float* pc1w  = (const float*)d_in[29];
    const float* pc1b  = (const float*)d_in[30];
    const float* pbn1g = (const float*)d_in[31];
    const float* pbn1b = (const float*)d_in[32];
    const float* pc2w  = (const float*)d_in[33];
    const float* pc2b  = (const float*)d_in[34];
    const float* pbn2g = (const float*)d_in[35];
    const float* pbn2b = (const float*)d_in[36];
    const float* capsw = (const float*)d_in[37];
    const float* fc1w  = (const float*)d_in[38];
    const float* fc1b  = (const float*)d_in[39];
    const float* fc2w  = (const float*)d_in[40];
    const float* fc2b  = (const float*)d_in[41];
    const float* fc3w  = (const float*)d_in[42];
    const float* fc3b  = (const float*)d_in[43];

    const int B = in_sizes[2] / (23 * 34);   // 4096
    const int BT = B / 16;

    size_t off = 0;
    auto alloc = [&](size_t bytes) -> void* {
        void* p = (char*)d_ws + off;
        off += (bytes + 255) & ~(size_t)255;
        return p;
    };

    // im2col scratch (reused by every conv stage; conv2 is the largest)
    f16* imbuf = (f16*)alloc((size_t)27 * B * 384 * 2);
    // packed weights
    f16* wp1  = (f16*)alloc((size_t)64  * 96  * 2);
    f16* wp12 = (f16*)alloc((size_t)128 * 256 * 2);
    f16* wp2  = (f16*)alloc((size_t)256 * 384 * 2);
    f16* wp3  = (f16*)alloc((size_t)32  * 768 * 2);
    f16* wpp0 = (f16*)alloc((size_t)32  * 32  * 2);
    f16* wpc1 = (f16*)alloc((size_t)64  * 96  * 2);
    f16* wpc2 = (f16*)alloc((size_t)32  * 192 * 2);
    f16* cwt  = (f16*)alloc((size_t)40 * 224 * 32 * 2);
    f16* f1t  = (f16*)alloc((size_t)224 * 224 * 2);
    f16* f2t  = (f16*)alloc((size_t)224 * 224 * 2);
    // activations (all f16)
    f16* a1   = (f16*)alloc((size_t)B * 64 * 32 * 2);
    f16* a2   = (f16*)alloc((size_t)B * 128 * 29 * 2);
    f16* a2a  = (f16*)alloc((size_t)B * 128 * 29 * 2);
    f16* a3   = (f16*)alloc((size_t)B * 256 * 27 * 2);
    f16* a3p  = (f16*)alloc((size_t)B * 256 * 13 * 2);
    f16* u    = (f16*)alloc((size_t)B * 40 * 32 * 2);   // K-padded capsule input
    f16* zp0  = (f16*)alloc((size_t)B * 32 * 15 * 2);
    f16* zp0a = (f16*)alloc((size_t)B * 32 * 15 * 2);
    f16* zp1  = (f16*)alloc((size_t)B * 64 * 13 * 2);
    f16* up   = (f16*)alloc((size_t)B * 40 * 224 * 2);
    f16* ctx  = (f16*)alloc((size_t)B * 224 * 2);
    f16* h1   = (f16*)alloc((size_t)B * 224 * 2);
    f16* h2   = (f16*)alloc((size_t)B * 224 * 2);

    auto g1 = [](int n) { return dim3((n + 255) / 256); };

    // --- pack weights ---
    k_pack_w<<<g1(64 * 96), 256, 0, stream>>>(c1w,  wp1,  64, 69,  64, 96);
    k_pack_w<<<g1(128 * 256), 256, 0, stream>>>(c12w, wp12, 128, 256, 128, 256);
    k_pack_w<<<g1(256 * 384), 256, 0, stream>>>(c2w,  wp2,  256, 384, 256, 384);
    k_pack_w<<<g1(32 * 768), 256, 0, stream>>>(c3w,  wp3,  20, 768, 32, 768);
    k_pack_w<<<g1(32 * 32), 256, 0, stream>>>(p0w,  wpp0, 32, 23, 32, 32);
    k_pack_w<<<g1(64 * 96), 256, 0, stream>>>(pc1w, wpc1, 64, 96, 64, 96);
    k_pack_w<<<g1(32 * 192), 256, 0, stream>>>(pc2w, wpc2, 20, 192, 32, 192);
    k_pack_caps<<<g1(40 * 224 * 32), 256, 0, stream>>>(capsw, cwt);
    k_pack_wt<<<g1(224 * 224), 256, 0, stream>>>(fc1w, f1t, 220, 200, 224, 224);
    k_pack_wt<<<g1(224 * 224), 256, 0, stream>>>(fc2w, f2t, 200, 200, 224, 224);
    k_zero16<<<g1(B * 40 * 32), 256, 0, stream>>>(u, B * 40 * 32);

    // --- HLA branch ---
    k_im2col_f32<<<g1(32 * B * 96), 256, 0, stream>>>(x2, imbuf, B, 23, 34, 3, 32, 69, 96);
    k_gemm16<96, 4, 2><<<dim3(BT, 1, 32), 32, 0, stream>>>(
        imbuf, (long long)B * 96, 96, wp1, 0, c1b, bn1g, bn1b,
        a1, 1, 64 * 32, 32, 0, 64);

    k_im2col_f16<<<g1(29 * B * 256), 256, 0, stream>>>(a1, imbuf, B, 64, 32, 4, 29, 256, 256);
    k_gemm16<256, 4, 2><<<dim3(BT, 2, 29), 32, 0, stream>>>(
        imbuf, (long long)B * 256, 256, wp12, 0, c12b, bn12g, bn12b,
        a2, 1, 128 * 29, 29, 0, 128);

    k_attention<<<B, 32, 0, stream>>>(a2, atthw, atthb, a2a, 128, 29);

    k_im2col_f16<<<g1(27 * B * 384), 256, 0, stream>>>(a2a, imbuf, B, 128, 29, 3, 27, 384, 384);
    k_gemm16<384, 4, 2><<<dim3(BT, 4, 27), 32, 0, stream>>>(
        imbuf, (long long)B * 384, 384, wp2, 0, c2b, bn2g, bn2b,
        a3, 1, 256 * 27, 27, 0, 256);

    {
        int n = B * 256 * 13;
        k_maxpool<<<g1(n), 256, 0, stream>>>(a3, a3p, n, 27, 13);
    }

    k_im2col_f16<<<g1(11 * B * 768), 256, 0, stream>>>(a3p, imbuf, B, 256, 13, 3, 11, 768, 768);
    k_gemm16<768, 2, 2><<<dim3(BT, 1, 11), 32, 0, stream>>>(
        imbuf, (long long)B * 768, 768, wp3, 0, c3b, bn3g, bn3b,
        u, 1, 40 * 32, 32, 0, 20);                       // channels 0..19 of u

    // --- Peptide branch ---
    k_im2col_f32<<<g1(15 * B * 32), 256, 0, stream>>>(pep, imbuf, B, 23, 15, 1, 15, 23, 32);
    k_gemm16<32, 2, 0><<<dim3(BT, 1, 15), 32, 0, stream>>>(
        imbuf, (long long)B * 32, 32, wpp0, 0, nullptr, nullptr, nullptr,
        zp0, 1, 32 * 15, 15, 0, 32);

    k_attention<<<B, 32, 0, stream>>>(zp0, attpw, attpb, zp0a, 32, 15);

    k_im2col_f16<<<g1(13 * B * 96), 256, 0, stream>>>(zp0a, imbuf, B, 32, 15, 3, 13, 96, 96);
    k_gemm16<96, 4, 2><<<dim3(BT, 1, 13), 32, 0, stream>>>(
        imbuf, (long long)B * 96, 96, wpc1, 0, pc1b, pbn1g, pbn1b,
        zp1, 1, 64 * 13, 13, 0, 64);

    k_im2col_f16<<<g1(11 * B * 192), 256, 0, stream>>>(zp1, imbuf, B, 64, 13, 3, 11, 192, 192);
    k_gemm16<192, 2, 2><<<dim3(BT, 1, 11), 32, 0, stream>>>(
        imbuf, (long long)B * 192, 192, wpc2, 0, pc2b, pbn2g, pbn2b,
        u, 1, 40 * 32, 32, 20 * 32, 20);                 // channels 20..39 of u

    // --- Capsule projection: 40 batched GEMMs, K padded 11->32 ---
    k_gemm16<32, 2, 0><<<dim3(BT, 7, 40), 32, 0, stream>>>(
        u, 32, 40 * 32, cwt, (long long)224 * 32, nullptr, nullptr, nullptr,
        up, 224, 40 * 224, 1, 0, 220);

    // --- Dynamic routing ---
    k_routing<<<B, 256, 0, stream>>>(up, ctx);

    // --- Predictor ---
    k_gemm16<224, 2, 1><<<dim3(BT, 7, 1), 32, 0, stream>>>(
        ctx, 0, 224, f1t, 0, fc1b, nullptr, nullptr,
        h1, 0, 224, 1, 0, 200);                          // zero-pads cols 200..223
    k_gemm16<224, 2, 1><<<dim3(BT, 7, 1), 32, 0, stream>>>(
        h1, 0, 224, f2t, 0, fc2b, nullptr, nullptr,
        h2, 0, 224, 1, 0, 200);
    k_fc3<<<g1(B), 256, 0, stream>>>(h2, fc3w, fc3b, (float*)d_out, B);
}
